// DeepSpeedSelfAttention_24446953848858
// MI455X (gfx1250) — compile-verified, matrix operand order checked by reading, not run on hardware
//
#include <hip/hip_runtime.h>

typedef __attribute__((ext_vector_type(16))) __bf16 v16bf;
typedef __attribute__((ext_vector_type(8)))  float  v8f;

// ---------------------------------------------------------------- helpers

__device__ __forceinline__ unsigned short f2bf(float f) {
    unsigned int u = __float_as_uint(f);
    unsigned int r = (u + 0x7FFFu + ((u >> 16) & 1u)) >> 16;   // round-nearest-even
    return (unsigned short)r;
}

__device__ __forceinline__ v8f zero8() {
    v8f z = {0.f, 0.f, 0.f, 0.f, 0.f, 0.f, 0.f, 0.f};
    return z;
}

// Load a 16x32 bf16 WMMA fragment (A layout; B uses the same per-lane map with
// the "row" meaning the N index) from a row-major bf16 (ushort) matrix.
// ISA layout: lane<16 -> row=lane,    K elems {0..7, 16..23}
//             lane>=16 -> row=lane-16, K elems {8..15, 24..31}
__device__ __forceinline__ v16bf load_frag(const unsigned short* base, int stride, int lane) {
    const int r    = lane & 15;
    const int koff = (lane >> 4) << 3;            // 0 or 8
    const unsigned short* p0 = base + (size_t)r * stride + koff;
    union { uint4 q[2]; unsigned int u[8]; v16bf v; } t;
    t.q[0] = *(const uint4*)(p0);                 // K koff..koff+7
    t.q[1] = *(const uint4*)(p0 + 16);            // K 16+koff..16+koff+7
    return t.v;
}

__device__ __forceinline__ v8f wmma_bf16(v16bf a, v16bf b, v8f c) {
    return __builtin_amdgcn_wmma_f32_16x16x32_bf16(false, a, false, b, (short)0, c, false, false);
}

// wave-local LDS write->read ordering: LDS ops from one wave complete in order
// (DScnt, ISA ch.11); wait DScnt==0 + compiler memory barrier is sufficient.
__device__ __forceinline__ void lds_wave_fence() {
    asm volatile("s_wait_dscnt 0x0" ::: "memory");
}

// reductions across the 16-lane half-wave that owns one score row group
__device__ __forceinline__ float half_max(float v) {
    #pragma unroll
    for (int m = 1; m < 16; m <<= 1) v = fmaxf(v, __shfl_xor(v, m, 32));
    return v;
}
__device__ __forceinline__ float half_sum(float v) {
    #pragma unroll
    for (int m = 1; m < 16; m <<= 1) v += __shfl_xor(v, m, 32);
    return v;
}

// ---------------------------------------------------------------- kernels

// elementwise fp32 -> bf16
__global__ void __launch_bounds__(256) cvt_bf16_kernel(const float* __restrict__ in,
                                                       unsigned short* __restrict__ out, int n) {
    int i = blockIdx.x * 256 + threadIdx.x;
    if (i < n) out[i] = f2bf(in[i]);
}

// LayerNorm over H=1024 per token; one block per token, 4 elems/thread
__global__ void __launch_bounds__(256) ln_kernel(const float* __restrict__ x,
                                                 const float* __restrict__ w,
                                                 const float* __restrict__ bb,
                                                 unsigned short* __restrict__ o) {
    __shared__ float ssum[8], ssq[8];
    const int row = blockIdx.x;
    const float4 xv = ((const float4*)(x + (size_t)row * 1024))[threadIdx.x];
    float s = xv.x + xv.y + xv.z + xv.w;
    float q = xv.x * xv.x + xv.y * xv.y + xv.z * xv.z + xv.w * xv.w;
    #pragma unroll
    for (int m = 1; m < 32; m <<= 1) { s += __shfl_xor(s, m, 32); q += __shfl_xor(q, m, 32); }
    if ((threadIdx.x & 31) == 0) { ssum[threadIdx.x >> 5] = s; ssq[threadIdx.x >> 5] = q; }
    __syncthreads();
    float tot = 0.f, tq = 0.f;
    #pragma unroll
    for (int i = 0; i < 8; ++i) { tot += ssum[i]; tq += ssq[i]; }
    const float mu   = tot * (1.0f / 1024.0f);
    const float var  = tq * (1.0f / 1024.0f) - mu * mu;
    const float rstd = rsqrtf(var + 1e-12f);
    const float4 wv = ((const float4*)w)[threadIdx.x];
    const float4 bv = ((const float4*)bb)[threadIdx.x];
    ushort4 rr;
    rr.x = f2bf((xv.x - mu) * rstd * wv.x + bv.x);
    rr.y = f2bf((xv.y - mu) * rstd * wv.y + bv.y);
    rr.z = f2bf((xv.z - mu) * rstd * wv.z + bv.z);
    rr.w = f2bf((xv.w - mu) * rstd * wv.w + bv.w);
    ((ushort4*)(o + (size_t)row * 1024))[threadIdx.x] = rr;
}

// QKV GEMM: [8192,1024]bf16 x [3072,1024]^T bf16 -> fp32 + bias, scattered epilogue.
// Each wave: 32x64 output tile; per K-step 2 A-frags x 4 B-frags -> 8 WMMAs,
// so each B fragment is reused twice and each A fragment 4x.
__global__ void __launch_bounds__(256) qkv_gemm_kernel(
    const unsigned short* __restrict__ A, const unsigned short* __restrict__ W,
    const float* __restrict__ bias,
    float* __restrict__ key_out, float* __restrict__ val_out,
    unsigned short* __restrict__ qa, unsigned short* __restrict__ ka,
    unsigned short* __restrict__ vt) {
    const int lane = threadIdx.x & 31;
    const int wid  = blockIdx.x * 8 + (threadIdx.x >> 5);
    const int m0   = (wid / 48) * 32;
    const int n0   = (wid % 48) * 64;
    v8f acc[2][4];
    #pragma unroll
    for (int r = 0; r < 2; ++r)
        #pragma unroll
        for (int t = 0; t < 4; ++t) acc[r][t] = zero8();
    const unsigned short* Abase0 = A + (size_t)m0 * 1024;
    const unsigned short* Abase1 = A + (size_t)(m0 + 16) * 1024;
    for (int kk = 0; kk < 1024; kk += 32) {
        __builtin_prefetch(Abase0 + kk + 32, 0, 3);
        __builtin_prefetch(Abase1 + kk + 32, 0, 3);
        v16bf a0 = load_frag(Abase0 + kk, 1024, lane);
        v16bf a1 = load_frag(Abase1 + kk, 1024, lane);
        #pragma unroll
        for (int t = 0; t < 4; ++t) {
            v16bf b = load_frag(W + (size_t)(n0 + t * 16) * 1024 + kk, 1024, lane);
            acc[0][t] = wmma_bf16(a0, b, acc[0][t]);
            acc[1][t] = wmma_bf16(a1, b, acc[1][t]);
        }
    }
    const int hi = lane >> 4, col = lane & 15;
    const int region = n0 >> 10;          // 0=q 1=k 2=v (64-col strips never straddle)
    const int nl0 = n0 & 1023;
    #pragma unroll
    for (int r = 0; r < 2; ++r) {
        #pragma unroll
        for (int j = 0; j < 8; ++j) {
            const int m = m0 + r * 16 + j + 8 * hi;
            const int bidx = m >> 11, srow = m & 2047;
            #pragma unroll
            for (int t = 0; t < 4; ++t) {
                const int nl = nl0 + t * 16 + col;
                const float v = acc[r][t][j] + bias[region * 1024 + nl];
                const int hh = nl >> 6, dd = nl & 63;
                const size_t hoff = ((size_t)(bidx * 16 + hh) * 2048 + srow) * 64 + dd;
                if (region == 0) {
                    qa[hoff] = f2bf(v * 0.125f);                     // fold 1/sqrt(64) into Q
                } else if (region == 1) {
                    key_out[(size_t)m * 1024 + nl] = v;
                    ka[hoff] = f2bf(v);
                } else {
                    val_out[(size_t)m * 1024 + nl] = v;
                    vt[((size_t)(bidx * 16 + hh) * 64 + dd) * 2048 + srow] = f2bf(v); // V^T
                }
            }
        }
    }
}

// Flash attention: wave = 16 queries of one (b,h); stream 32 keys/iter, online softmax.
// P tile is wave-private in LDS -> only wave-local DScnt ordering needed, no barriers.
__global__ void __launch_bounds__(256) attn_kernel(
    const unsigned short* __restrict__ qa, const unsigned short* __restrict__ ka,
    const unsigned short* __restrict__ vt, const float* __restrict__ mask,
    float* __restrict__ ctx_out, unsigned short* __restrict__ ctx_bf) {
    __shared__ __align__(16) unsigned short psh[8 * 16 * 32];   // 1KB P tile per wave
    const int lane = threadIdx.x & 31;
    const int warp = threadIdx.x >> 5;
    const int wid  = blockIdx.x * 8 + warp;
    const int qt = wid & 127;
    const int bh = wid >> 7;
    const int b = bh >> 4, h = bh & 15;
    const int m0 = qt * 16;
    const int hi = lane >> 4, col = lane & 15;
    unsigned short* pw = psh + warp * (16 * 32);

    const unsigned short* qbase = qa + (size_t)(bh * 2048 + m0) * 64;
    const v16bf aq0 = load_frag(qbase,      64, lane);
    const v16bf aq1 = load_frag(qbase + 32, 64, lane);

    float mrow[8], lrow[8];
    v8f acc[4];
    #pragma unroll
    for (int j = 0; j < 8; ++j) { mrow[j] = -1e30f; lrow[j] = 0.f; }
    #pragma unroll
    for (int t = 0; t < 4; ++t) acc[t] = zero8();

    for (int kb = 0; kb < 2048; kb += 32) {
        const unsigned short* kbase = ka + (size_t)(bh * 2048 + kb) * 64;
        v8f sc0 = zero8(), sc1 = zero8();
        sc0 = wmma_bf16(aq0, load_frag(kbase,                64, lane), sc0);
        sc0 = wmma_bf16(aq1, load_frag(kbase + 32,           64, lane), sc0);
        sc1 = wmma_bf16(aq0, load_frag(kbase + 16 * 64,      64, lane), sc1);
        sc1 = wmma_bf16(aq1, load_frag(kbase + 16 * 64 + 32, 64, lane), sc1);

        const float mk0 = mask[b * 2048 + kb + col];
        const float mk1 = mask[b * 2048 + kb + 16 + col];

        #pragma unroll
        for (int j = 0; j < 8; ++j) {
            const float s0 = sc0[j] + mk0;
            const float s1 = sc1[j] + mk1;
            const float rm = half_max(fmaxf(s0, s1));
            const float mnew  = fmaxf(mrow[j], rm);
            const float alpha = __expf(mrow[j] - mnew);
            const float p0 = __expf(s0 - mnew);
            const float p1 = __expf(s1 - mnew);
            lrow[j] = lrow[j] * alpha + half_sum(p0 + p1);
            mrow[j] = mnew;
            #pragma unroll
            for (int t = 0; t < 4; ++t) acc[t][j] *= alpha;
            const int row = j + 8 * hi;
            pw[row * 32 + col]      = f2bf(p0);
            pw[row * 32 + 16 + col] = f2bf(p1);
        }
        lds_wave_fence();                       // stores -> loads, same wave, in order
        const v16bf pa = load_frag(pw, 32, lane);
        #pragma unroll
        for (int t = 0; t < 4; ++t) {
            const unsigned short* vb = vt + (size_t)(bh * 64 + t * 16) * 2048 + kb;
            acc[t] = wmma_bf16(pa, load_frag(vb, 2048, lane), acc[t]);
        }
        lds_wave_fence();                       // loads done before next overwrite
    }

    #pragma unroll
    for (int j = 0; j < 8; ++j) {
        const float rinv = 1.0f / lrow[j];
        const int srow = m0 + j + 8 * hi;
        const size_t rowoff = ((size_t)b * 2048 + srow) * 1024 + h * 64;
        #pragma unroll
        for (int t = 0; t < 4; ++t) {
            const float v = acc[t][j] * rinv;
            ctx_out[rowoff + t * 16 + col] = v;     // context_layer [B,S,H]
            ctx_bf[rowoff + t * 16 + col]  = f2bf(v);
        }
    }
}

// Output projection: [8192,1024]bf16 x [1024,1024]^T bf16 -> fp32, 32x64 wave tile
__global__ void __launch_bounds__(256) out_gemm_kernel(
    const unsigned short* __restrict__ A, const unsigned short* __restrict__ W,
    float* __restrict__ out) {
    const int lane = threadIdx.x & 31;
    const int wid  = blockIdx.x * 8 + (threadIdx.x >> 5);
    const int m0   = (wid / 16) * 32;
    const int n0   = (wid % 16) * 64;
    v8f acc[2][4];
    #pragma unroll
    for (int r = 0; r < 2; ++r)
        #pragma unroll
        for (int t = 0; t < 4; ++t) acc[r][t] = zero8();
    const unsigned short* Abase0 = A + (size_t)m0 * 1024;
    const unsigned short* Abase1 = A + (size_t)(m0 + 16) * 1024;
    for (int kk = 0; kk < 1024; kk += 32) {
        __builtin_prefetch(Abase0 + kk + 32, 0, 3);
        __builtin_prefetch(Abase1 + kk + 32, 0, 3);
        v16bf a0 = load_frag(Abase0 + kk, 1024, lane);
        v16bf a1 = load_frag(Abase1 + kk, 1024, lane);
        #pragma unroll
        for (int t = 0; t < 4; ++t) {
            v16bf b = load_frag(W + (size_t)(n0 + t * 16) * 1024 + kk, 1024, lane);
            acc[0][t] = wmma_bf16(a0, b, acc[0][t]);
            acc[1][t] = wmma_bf16(a1, b, acc[1][t]);
        }
    }
    const int hi = lane >> 4, col = lane & 15;
    #pragma unroll
    for (int r = 0; r < 2; ++r)
        #pragma unroll
        for (int j = 0; j < 8; ++j) {
            const int m = m0 + r * 16 + j + 8 * hi;
            #pragma unroll
            for (int t = 0; t < 4; ++t)
                out[(size_t)m * 1024 + n0 + t * 16 + col] = acc[r][t][j];
        }
}

// ---------------------------------------------------------------- launcher

extern "C" void kernel_launch(void* const* d_in, const int* in_sizes, int n_in,
                              void* d_out, int out_size, void* d_ws, size_t ws_size,
                              hipStream_t stream) {
    (void)in_sizes; (void)n_in; (void)out_size; (void)ws_size;
    const float* input  = (const float*)d_in[0];
    const float* mask   = (const float*)d_in[1];
    const float* norm_w = (const float*)d_in[2];
    const float* norm_b = (const float*)d_in[3];
    const float* qkvw   = (const float*)d_in[4];
    const float* qkvb   = (const float*)d_in[5];
    const float* ow     = (const float*)d_in[6];

    float* out     = (float*)d_out;            // [8192,1024]
    float* key_out = out + 8388608;            // key_layer
    float* val_out = out + 16777216;           // value_layer
    float* ctx_out = out + 25165824;           // context_layer

    char* w = (char*)d_ws;
    unsigned short* ln_bf   = (unsigned short*)(w);              // 16 MB
    unsigned short* wqkv_bf = (unsigned short*)(w + 16777216);   //  6 MB
    unsigned short* ow_bf   = (unsigned short*)(w + 23068672);   //  2 MB
    unsigned short* qa      = (unsigned short*)(w + 25165824);   // 16 MB [b,h,s,d] bf16 (pre-scaled)
    unsigned short* ka      = (unsigned short*)(w + 41943040);   // 16 MB [b,h,s,d] bf16
    unsigned short* vt      = (unsigned short*)(w + 58720256);   // 16 MB [b,h,d,s] bf16
    unsigned short* ctx_bf  = (unsigned short*)(w + 75497472);   // 16 MB

    cvt_bf16_kernel<<<12288, 256, 0, stream>>>(qkvw, wqkv_bf, 3145728);
    cvt_bf16_kernel<<<4096,  256, 0, stream>>>(ow,   ow_bf,   1048576);
    ln_kernel<<<8192, 256, 0, stream>>>(input, norm_w, norm_b, ln_bf);
    // 256 M-tiles(32) x 48 N-strips(64) = 12288 wave tiles / 8 waves = 1536 blocks
    qkv_gemm_kernel<<<1536, 256, 0, stream>>>(ln_bf, wqkv_bf, qkvb,
                                              key_out, val_out, qa, ka, vt);
    attn_kernel<<<1024, 256, 0, stream>>>(qa, ka, vt, mask, ctx_out, ctx_bf);
    // 256 M-tiles(32) x 16 N-strips(64) = 4096 wave tiles / 8 waves = 512 blocks
    out_gemm_kernel<<<512, 256, 0, stream>>>(ctx_bf, ow_bf, out);
}
